// MambaBlock_z_74646531604856
// MI455X (gfx1250) — matrix operand order repaired and checked by reading, not compiled, not run on hardware
//
#include <hip/hip_runtime.h>
#include <hip/hip_bf16.h>

// ---------------- problem constants ----------------
#define BQ   4
#define LQ   2048
#define DQ   768
#define DIQ  1536
#define DSQ  16
#define DRQ  48
#define MQ   (BQ * LQ)        // 8192 token rows
#define NXZ  (2 * DIQ)        // 3072
#define NDBL (DRQ + 2 * DSQ)  // 80

// LDS tile geometry for TDM GEMMs: 128 rows x 32 bf16, padded to 80B pitch
// (TDM pad: 16 DWORDs data + 4 DWORDs pad) -> bank-conflict-free frag reads.
#define LROW 40               // shorts per LDS row (32 data + 8 pad)
#define TROWS 128
#define TBYTES (TROWS * LROW) // shorts per staged tile

// ---------------- types ----------------
typedef __attribute__((ext_vector_type(16))) __bf16   v16bf;
typedef __attribute__((ext_vector_type(8)))  float    v8f;
typedef __attribute__((ext_vector_type(4)))  unsigned u32x4;
typedef __attribute__((ext_vector_type(8)))  int      i32x8;
typedef __attribute__((ext_vector_type(4)))  int      i32x4;

#if __has_include(<hip/amd_detail/amd_gfx1250_TDM.h>)
#define TDM_SIX_ARGS 1
#endif

__device__ __forceinline__ unsigned short f2bf(float f) {
    union { float f; unsigned u; } cv; cv.f = f;
    unsigned r = cv.u + 0x7FFFu + ((cv.u >> 16) & 1u);
    return (unsigned short)(r >> 16);
}

__device__ __forceinline__ float silu_f(float v) {
    return v / (1.f + __expf(-v));
}

// ---- TDM: async DMA of a (rows x 32) bf16 tile, row-major stride lda elems,
// into LDS at lds_off with 16B pad after every 64B row (ISA ch.8 D# layout).
__device__ __forceinline__ void tdm_load_tile(unsigned lds_off,
                                              unsigned long long gaddr,
                                              int lda, int rows) {
    u32x4 g0;
    g0.x = 1u;                                            // count=1, user desc
    g0.y = lds_off;                                       // LDS byte address
    g0.z = (unsigned)(gaddr & 0xFFFFFFFFu);               // global_addr[31:0]
    g0.w = (unsigned)((gaddr >> 32) & 0x1FFFFFFu)         // global_addr[56:32]
         | (2u << 30);                                    // type=2 ("image")
    i32x8 g1;
    g1[0] = (int)(0x00010000u                             // data_size=1 (2B)
         | (1u << 20)                                     // pad_enable
         | (3u << 22)                                     // pad_interval: 16 DW
         | (3u << 25));                                   // pad_amount: 4 DW
    g1[1] = (int)(32u << 16);                             // tensor_dim0 lo16=32
    g1[2] = (int)(((unsigned)rows & 0xFFFFu) << 16);      // tensor_dim1 lo16
    g1[3] = (int)(32u << 16);                             // tile_dim0 = 32
    g1[4] = rows;                                         // tile_dim1 = rows
    g1[5] = lda;                                          // dim0_stride[31:0]
    g1[6] = 0;
    g1[7] = 0;
    i32x4 gz = {0, 0, 0, 0};
#ifdef TDM_SIX_ARGS
    i32x8 gz8 = {0, 0, 0, 0, 0, 0, 0, 0};
    __builtin_amdgcn_tensor_load_to_lds(g0, g1, gz, gz, gz8, 0);
#else
    __builtin_amdgcn_tensor_load_to_lds(g0, g1, gz, gz, 0);
#endif
}

// ---- LDS fragment loaders (padded 80B pitch) ----
__device__ __forceinline__ v16bf lds_a_frag(const unsigned short* sA, int row0, int lane) {
    int m  = lane & 15;
    int kb = (lane >> 4) << 3;
    const unsigned short* p = sA + (row0 + m) * LROW + kb;
    union { u32x4 u[2]; v16bf v; } f;
    f.u[0] = *(const u32x4*)(p);
    f.u[1] = *(const u32x4*)(p + 16);
    return f.v;
}
__device__ __forceinline__ v16bf lds_b_frag(const unsigned short* sB, int n0, int lane) {
    int n  = lane & 15;
    int kb = (lane >> 4) << 3;
    const unsigned short* p = sB + (n0 + n) * LROW + kb;
    union { u32x4 u[2]; v16bf v; } f;
    f.u[0] = *(const u32x4*)(p);
    f.u[1] = *(const u32x4*)(p + 16);
    return f.v;
}

// ---- direct-global fragment loaders (for the small N=80 GEMM) ----
__device__ __forceinline__ v16bf load_a_frag(const unsigned short* __restrict__ A,
                                             int lda, int row0, int k0, int lane) {
    int m  = lane & 15;
    int kb = (lane >> 4) << 3;
    const unsigned short* p = A + (size_t)(row0 + m) * lda + k0 + kb;
    union { u32x4 u[2]; v16bf v; } f;
    f.u[0] = *(const u32x4*)(p);
    f.u[1] = *(const u32x4*)(p + 16);
    return f.v;
}
__device__ __forceinline__ v16bf load_b_frag(const unsigned short* __restrict__ W,
                                             int ldw, int n0, int k0, int lane) {
    int n  = lane & 15;
    int kb = (lane >> 4) << 3;
    const unsigned short* p = W + (size_t)(n0 + n) * ldw + k0 + kb;
    union { u32x4 u[2]; v16bf v; } f;
    f.u[0] = *(const u32x4*)(p);
    f.u[1] = *(const u32x4*)(p + 16);
    return f.v;
}

#define WMMA_BF16(A, B, C) \
    __builtin_amdgcn_wmma_f32_16x16x32_bf16(false, (A), false, (B), (short)0, (C), false, false)

// ---------------- elementwise f32 -> bf16 ----------------
__global__ __launch_bounds__(256) void k_cvt_bf16(const float* __restrict__ in,
                                                  unsigned short* __restrict__ out, int n) {
    int i = blockIdx.x * 256 + threadIdx.x;
    if (i < n) out[i] = f2bf(in[i]);
}

// ---------------- ada = silu(c) @ ada_w^T + ada_b ----------------
__global__ __launch_bounds__(256) void k_ada(const float* __restrict__ c,
                                             const float* __restrict__ w,
                                             const float* __restrict__ bias,
                                             float* __restrict__ ada) {
    __shared__ float sc[2 * DQ];
    int b = blockIdx.x / 9;
    int j = (blockIdx.x % 9) * 256 + threadIdx.x;
    for (int k = threadIdx.x; k < 2 * DQ; k += 256)
        sc[k] = silu_f(c[b * 2 * DQ + k]);
    __syncthreads();
    float acc = bias[j];
    const float* wr = w + (size_t)j * (2 * DQ);
    for (int k = 0; k < 2 * DQ; ++k) acc += sc[k] * wr[k];
    ada[b * 3 * DQ + j] = acc;
}

// ---------------- LayerNorm + AdaLN modulate -> xm (bf16) ----------------
__global__ __launch_bounds__(256) void k_ln_mod(const float* __restrict__ x,
                                                const float* __restrict__ lnw,
                                                const float* __restrict__ lnb,
                                                const float* __restrict__ ada,
                                                unsigned short* __restrict__ xm) {
    int r = blockIdx.x;
    int b = r >> 11;
    const float* xr = x + (size_t)r * DQ;
    float xv[3], s = 0.f, q = 0.f;
#pragma unroll
    for (int i = 0; i < 3; ++i) {
        float v = xr[threadIdx.x + i * 256];
        xv[i] = v; s += v; q += v * v;
    }
#pragma unroll
    for (int off = 16; off > 0; off >>= 1) {
        s += __shfl_down(s, off, 32);
        q += __shfl_down(q, off, 32);
    }
    __shared__ float red[16];
    __shared__ float stats[2];
    int wid = threadIdx.x >> 5;
    if ((threadIdx.x & 31) == 0) { red[wid] = s; red[8 + wid] = q; }
    __syncthreads();
    if (threadIdx.x == 0) {
        float ts = 0.f, tq = 0.f;
        for (int i = 0; i < 8; ++i) { ts += red[i]; tq += red[8 + i]; }
        float mu = ts * (1.f / DQ);
        stats[0] = mu;
        stats[1] = rsqrtf(tq * (1.f / DQ) - mu * mu + 1e-5f);
    }
    __syncthreads();
    float mu = stats[0], rstd = stats[1];
#pragma unroll
    for (int i = 0; i < 3; ++i) {
        int cc = threadIdx.x + i * 256;
        float xn    = (xv[i] - mu) * rstd * lnw[cc] + lnb[cc];
        float shift = ada[b * 3 * DQ + cc];
        float scale = ada[b * 3 * DQ + DQ + cc];
        xm[(size_t)r * DQ + cc] = f2bf(xn * (1.f + scale) + shift);
    }
}

// ---------------- GEMM1: xz[M,3072] = xm[M,768] @ in_proj_w^T ---------------
// TDM 3-stage pipeline: 128x128 block tile, 8 waves of 32(M)x64(N).
__global__ __launch_bounds__(256) void k_gemm_xz(const unsigned short* __restrict__ a,
                                                 const unsigned short* __restrict__ w,
                                                 float* __restrict__ xz) {
    __shared__ unsigned short sA[3][TBYTES];
    __shared__ unsigned short sB[3][TBYTES];
    int lane  = threadIdx.x & 31;
    int wave  = threadIdx.x >> 5;
    int waveM = wave >> 1;          // 0..3
    int nOff  = (wave & 1) * 64;    // 0 or 64
    unsigned long long aBase =
        (unsigned long long)(uintptr_t)a + (unsigned long long)blockIdx.y * 128ull * DQ * 2ull;
    unsigned long long wBase =
        (unsigned long long)(uintptr_t)w + (unsigned long long)blockIdx.x * 128ull * DQ * 2ull;
    const int KT = DQ / 32;         // 24
    if (wave == 0) {
        tdm_load_tile((unsigned)(uintptr_t)sA[0], aBase + 0ull, DQ, TROWS);
        tdm_load_tile((unsigned)(uintptr_t)sB[0], wBase + 0ull, DQ, TROWS);
        tdm_load_tile((unsigned)(uintptr_t)sA[1], aBase + 64ull, DQ, TROWS);
        tdm_load_tile((unsigned)(uintptr_t)sB[1], wBase + 64ull, DQ, TROWS);
    }
    v8f zero = {0.f,0.f,0.f,0.f,0.f,0.f,0.f,0.f};
    v8f c00 = zero, c01 = zero, c02 = zero, c03 = zero;
    v8f c10 = zero, c11 = zero, c12 = zero, c13 = zero;
    int cur = 0;
    for (int it = 0; it < KT; ++it) {
        if (wave == 0) {
            if (it + 1 < KT) __builtin_amdgcn_s_wait_tensorcnt(2);
            else             __builtin_amdgcn_s_wait_tensorcnt(0);
        }
        __syncthreads();
        if (wave == 0 && it + 2 < KT) {
            int nxt = cur + 2; if (nxt >= 3) nxt -= 3;
            unsigned long long koff = (unsigned long long)(it + 2) * 64ull;
            tdm_load_tile((unsigned)(uintptr_t)sA[nxt], aBase + koff, DQ, TROWS);
            tdm_load_tile((unsigned)(uintptr_t)sB[nxt], wBase + koff, DQ, TROWS);
        }
        const unsigned short* As = sA[cur];
        const unsigned short* Bs = sB[cur];
        v16bf a0 = lds_a_frag(As, waveM * 32,      lane);
        v16bf a1 = lds_a_frag(As, waveM * 32 + 16, lane);
        v16bf b0 = lds_b_frag(Bs, nOff +  0, lane);
        v16bf b1 = lds_b_frag(Bs, nOff + 16, lane);
        v16bf b2 = lds_b_frag(Bs, nOff + 32, lane);
        v16bf b3 = lds_b_frag(Bs, nOff + 48, lane);
        c00 = WMMA_BF16(a0, b0, c00);
        c10 = WMMA_BF16(a1, b0, c10);
        c01 = WMMA_BF16(a0, b1, c01);
        c11 = WMMA_BF16(a1, b1, c11);
        c02 = WMMA_BF16(a0, b2, c02);
        c12 = WMMA_BF16(a1, b2, c12);
        c03 = WMMA_BF16(a0, b3, c03);
        c13 = WMMA_BF16(a1, b3, c13);
        cur = (cur + 1 == 3) ? 0 : cur + 1;
    }
    int rBase = blockIdx.y * 128 + waveM * 32 + ((lane >> 4) << 3);
    int cBase = blockIdx.x * 128 + nOff + (lane & 15);
#define EPI_XZ(ACC, SUB, NT)                                                       \
    _Pragma("unroll")                                                              \
    for (int i = 0; i < 8; ++i)                                                    \
        xz[(size_t)(rBase + (SUB) * 16 + i) * NXZ + cBase + (NT) * 16] = (ACC)[i];
    EPI_XZ(c00, 0, 0); EPI_XZ(c01, 0, 1); EPI_XZ(c02, 0, 2); EPI_XZ(c03, 0, 3);
    EPI_XZ(c10, 1, 0); EPI_XZ(c11, 1, 1); EPI_XZ(c12, 1, 2); EPI_XZ(c13, 1, 3);
#undef EPI_XZ
}

// ---------------- depthwise causal conv (K=4) + SiLU -> u (f32 + bf16) -------
__global__ __launch_bounds__(256) void k_conv(const float* __restrict__ xz,
                                              const float* __restrict__ cw,
                                              const float* __restrict__ cb,
                                              float* __restrict__ uf,
                                              unsigned short* __restrict__ ub) {
    int d = blockIdx.x * 256 + threadIdx.x;
    int r = blockIdx.y;
    int l = r & (LQ - 1);
    float acc = cb[d];
#pragma unroll
    for (int k = 0; k < 4; ++k) {
        int lp = l - 3 + k;
        if (lp >= 0) acc += cw[d * 4 + k] * xz[(size_t)(r - 3 + k) * NXZ + d];
    }
    float u = silu_f(acc);
    uf[(size_t)r * DIQ + d] = u;
    ub[(size_t)r * DIQ + d] = f2bf(u);
}

// ---------------- GEMM2: dbl[M,80] = u[M,1536] @ x_proj_w^T (direct WMMA) ----
__global__ __launch_bounds__(256) void k_gemm_dbl(const unsigned short* __restrict__ a,
                                                  const unsigned short* __restrict__ w,
                                                  float* __restrict__ dbl) {
    int lane  = threadIdx.x & 31;
    int wave  = threadIdx.x >> 5;
    int mBase = blockIdx.x * 128 + wave * 16;
    v8f zero = {0.f,0.f,0.f,0.f,0.f,0.f,0.f,0.f};
    v8f d0 = zero, d1 = zero, d2 = zero, d3 = zero, d4 = zero;
    for (int k0 = 0; k0 < DIQ; k0 += 32) {
        v16bf af = load_a_frag(a, DIQ, mBase, k0, lane);
        v16bf b0 = load_b_frag(w, DIQ,  0, k0, lane);
        v16bf b1 = load_b_frag(w, DIQ, 16, k0, lane);
        v16bf b2 = load_b_frag(w, DIQ, 32, k0, lane);
        v16bf b3 = load_b_frag(w, DIQ, 48, k0, lane);
        v16bf b4 = load_b_frag(w, DIQ, 64, k0, lane);
        d0 = WMMA_BF16(af, b0, d0);
        d1 = WMMA_BF16(af, b1, d1);
        d2 = WMMA_BF16(af, b2, d2);
        d3 = WMMA_BF16(af, b3, d3);
        d4 = WMMA_BF16(af, b4, d4);
    }
    int rlo = mBase + ((lane >> 4) << 3);
    int c0  = lane & 15;
#define EPI_DBL(ACC, NT)                                                   \
    _Pragma("unroll")                                                      \
    for (int i = 0; i < 8; ++i)                                            \
        dbl[(size_t)(rlo + i) * NDBL + (NT) * 16 + c0] = (ACC)[i];
    EPI_DBL(d0, 0); EPI_DBL(d1, 1); EPI_DBL(d2, 2); EPI_DBL(d3, 3); EPI_DBL(d4, 4);
#undef EPI_DBL
}

// ---------------- delta = softplus(dt @ dt_proj_w^T + b) ---------------------
__global__ __launch_bounds__(256) void k_delta(const float* __restrict__ dbl,
                                               const float* __restrict__ wt,
                                               const float* __restrict__ bt,
                                               float* __restrict__ delta) {
    __shared__ float s[DRQ];
    int r = blockIdx.y;
    int d = blockIdx.x * 256 + threadIdx.x;
    if (threadIdx.x < DRQ) s[threadIdx.x] = dbl[(size_t)r * NDBL + threadIdx.x];
    __syncthreads();
    float acc = bt[d];
    const float* wr = wt + (size_t)d * DRQ;
#pragma unroll
    for (int j = 0; j < DRQ; ++j) acc += s[j] * wr[j];
    delta[(size_t)r * DIQ + d] = (acc > 20.f) ? acc : log1pf(__expf(acc));
}

// ---------------- sequential SSM scan, fused gating -> yz (bf16) -------------
__global__ __launch_bounds__(256) void k_scan(const float* __restrict__ delta,
                                              const float* __restrict__ uf,
                                              const float* __restrict__ dbl,
                                              const float* __restrict__ A_log,
                                              const float* __restrict__ Dp,
                                              const float* __restrict__ xz,
                                              unsigned short* __restrict__ yz) {
    int b = blockIdx.x / 6;
    int d = (blockIdx.x % 6) * 256 + threadIdx.x;
    float A[DSQ], h[DSQ];
#pragma unroll
    for (int s = 0; s < DSQ; ++s) {
        A[s] = -__expf(A_log[d * DSQ + s]);
        h[s] = 0.f;
    }
    float Dv = Dp[d];
    __shared__ float sB[DSQ], sC[DSQ];
    for (int t = 0; t < LQ; ++t) {
        size_t row = (size_t)b * LQ + t;
        if (threadIdx.x < 32) {
            float v = dbl[row * NDBL + DRQ + threadIdx.x];
            if (threadIdx.x < DSQ) sB[threadIdx.x] = v;
            else                   sC[threadIdx.x - DSQ] = v;
        }
        __syncthreads();
        float dt = delta[row * DIQ + d];
        float ut = uf[row * DIQ + d];
        float du = dt * ut;
        float y = 0.f;
#pragma unroll
        for (int s = 0; s < DSQ; ++s) {
            float dA = __expf(dt * A[s]);
            h[s] = dA * h[s] + du * sB[s];
            y += h[s] * sC[s];
        }
        float yt = y + ut * Dv;
        float z  = xz[row * NXZ + DIQ + d];
        yz[row * DIQ + d] = f2bf(yt * silu_f(z));
        __syncthreads();
    }
}

// ---------------- GEMM3: out = x + gate * (yz @ out_proj_w^T), TDM -----------
__global__ __launch_bounds__(256) void k_gemm_out(const unsigned short* __restrict__ a,
                                                  const unsigned short* __restrict__ w,
                                                  const float* __restrict__ x,
                                                  const float* __restrict__ ada,
                                                  float* __restrict__ out) {
    __shared__ unsigned short sA[3][TBYTES];
    __shared__ unsigned short sB[3][TBYTES];
    int lane  = threadIdx.x & 31;
    int wave  = threadIdx.x >> 5;
    int waveM = wave >> 1;
    int nOff  = (wave & 1) * 64;
    unsigned long long aBase =
        (unsigned long long)(uintptr_t)a + (unsigned long long)blockIdx.y * 128ull * DIQ * 2ull;
    unsigned long long wBase =
        (unsigned long long)(uintptr_t)w + (unsigned long long)blockIdx.x * 128ull * DIQ * 2ull;
    const int KT = DIQ / 32;        // 48
    if (wave == 0) {
        tdm_load_tile((unsigned)(uintptr_t)sA[0], aBase + 0ull, DIQ, TROWS);
        tdm_load_tile((unsigned)(uintptr_t)sB[0], wBase + 0ull, DIQ, TROWS);
        tdm_load_tile((unsigned)(uintptr_t)sA[1], aBase + 64ull, DIQ, TROWS);
        tdm_load_tile((unsigned)(uintptr_t)sB[1], wBase + 64ull, DIQ, TROWS);
    }
    v8f zero = {0.f,0.f,0.f,0.f,0.f,0.f,0.f,0.f};
    v8f c00 = zero, c01 = zero, c02 = zero, c03 = zero;
    v8f c10 = zero, c11 = zero, c12 = zero, c13 = zero;
    int cur = 0;
    for (int it = 0; it < KT; ++it) {
        if (wave == 0) {
            if (it + 1 < KT) __builtin_amdgcn_s_wait_tensorcnt(2);
            else             __builtin_amdgcn_s_wait_tensorcnt(0);
        }
        __syncthreads();
        if (wave == 0 && it + 2 < KT) {
            int nxt = cur + 2; if (nxt >= 3) nxt -= 3;
            unsigned long long koff = (unsigned long long)(it + 2) * 64ull;
            tdm_load_tile((unsigned)(uintptr_t)sA[nxt], aBase + koff, DIQ, TROWS);
            tdm_load_tile((unsigned)(uintptr_t)sB[nxt], wBase + koff, DIQ, TROWS);
        }
        const unsigned short* As = sA[cur];
        const unsigned short* Bs = sB[cur];
        v16bf a0 = lds_a_frag(As, waveM * 32,      lane);
        v16bf a1 = lds_a_frag(As, waveM * 32 + 16, lane);
        v16bf b0 = lds_b_frag(Bs, nOff +  0, lane);
        v16bf b1 = lds_b_frag(Bs, nOff + 16, lane);
        v16bf b2 = lds_b_frag(Bs, nOff + 32, lane);
        v16bf b3 = lds_b_frag(Bs, nOff + 48, lane);
        c00 = WMMA_BF16(a0, b0, c00);
        c10 = WMMA_BF16(a1, b0, c10);
        c01 = WMMA_BF16(a0, b1, c01);
        c11 = WMMA_BF16(a1, b1, c11);
        c02 = WMMA_BF16(a0, b2, c02);
        c12 = WMMA_BF16(a1, b2, c12);
        c03 = WMMA_BF16(a0, b3, c03);
        c13 = WMMA_BF16(a1, b3, c13);
        cur = (cur + 1 == 3) ? 0 : cur + 1;
    }
    int rBase = blockIdx.y * 128 + waveM * 32 + ((lane >> 4) << 3);
    int cBase = blockIdx.x * 128 + nOff + (lane & 15);
#define EPI_OUT(ACC, SUB, NT)                                                    \
    _Pragma("unroll")                                                            \
    for (int i = 0; i < 8; ++i) {                                                \
        int r  = rBase + (SUB) * 16 + i;                                         \
        int bb = r >> 11;                                                        \
        int cc = cBase + (NT) * 16;                                              \
        out[(size_t)r * DQ + cc] =                                               \
            x[(size_t)r * DQ + cc] + ada[bb * 3 * DQ + 2 * DQ + cc] * (ACC)[i];  \
    }
    EPI_OUT(c00, 0, 0); EPI_OUT(c01, 0, 1); EPI_OUT(c02, 0, 2); EPI_OUT(c03, 0, 3);
    EPI_OUT(c10, 1, 0); EPI_OUT(c11, 1, 1); EPI_OUT(c12, 1, 2); EPI_OUT(c13, 1, 3);
#undef EPI_OUT
}

// ---------------- host launcher ----------------
extern "C" void kernel_launch(void* const* d_in, const int* in_sizes, int n_in,
                              void* d_out, int out_size, void* d_ws, size_t ws_size,
                              hipStream_t stream) {
    const float* x          = (const float*)d_in[0];
    const float* c          = (const float*)d_in[1];
    const float* ln_w       = (const float*)d_in[2];
    const float* ln_b       = (const float*)d_in[3];
    const float* ada_w      = (const float*)d_in[4];
    const float* ada_b      = (const float*)d_in[5];
    const float* in_proj_w  = (const float*)d_in[6];
    const float* conv_w     = (const float*)d_in[7];
    const float* conv_b     = (const float*)d_in[8];
    const float* x_proj_w   = (const float*)d_in[9];
    const float* dt_proj_w  = (const float*)d_in[10];
    const float* dt_proj_b  = (const float*)d_in[11];
    const float* A_log      = (const float*)d_in[12];
    const float* D_param    = (const float*)d_in[13];
    const float* out_proj_w = (const float*)d_in[14];
    float* out = (float*)d_out;

    char* base = (char*)d_ws;
    size_t off = 0;
    auto alloc = [&](size_t bytes) {
        size_t cur = off;
        off += (bytes + 255) & ~(size_t)255;
        return (void*)(base + cur);
    };
    float*          ws_ada   = (float*)         alloc((size_t)BQ * 3 * DQ * 4);
    unsigned short* ws_xm    = (unsigned short*)alloc((size_t)MQ * DQ * 2);
    unsigned short* ws_win   = (unsigned short*)alloc((size_t)NXZ * DQ * 2);
    unsigned short* ws_wxp   = (unsigned short*)alloc((size_t)NDBL * DIQ * 2);
    unsigned short* ws_wout  = (unsigned short*)alloc((size_t)DQ * DIQ * 2);
    float*          ws_xz    = (float*)         alloc((size_t)MQ * NXZ * 4);
    float*          ws_uf    = (float*)         alloc((size_t)MQ * DIQ * 4);
    unsigned short* ws_ub    = (unsigned short*)alloc((size_t)MQ * DIQ * 2);
    float*          ws_dbl   = (float*)         alloc((size_t)MQ * NDBL * 4);
    float*          ws_delta = (float*)         alloc((size_t)MQ * DIQ * 4);
    unsigned short* ws_yz    = (unsigned short*)alloc((size_t)MQ * DIQ * 2);

    // weight casts to bf16
    k_cvt_bf16<<<(NXZ * DQ + 255) / 256, 256, 0, stream>>>(in_proj_w, ws_win, NXZ * DQ);
    k_cvt_bf16<<<(NDBL * DIQ + 255) / 256, 256, 0, stream>>>(x_proj_w, ws_wxp, NDBL * DIQ);
    k_cvt_bf16<<<(DQ * DIQ + 255) / 256, 256, 0, stream>>>(out_proj_w, ws_wout, DQ * DIQ);

    // adaLN params
    k_ada<<<BQ * 9, 256, 0, stream>>>(c, ada_w, ada_b, ws_ada);

    // LN + modulate
    k_ln_mod<<<MQ, 256, 0, stream>>>(x, ln_w, ln_b, ws_ada, ws_xm);

    // in_proj GEMM (TDM + WMMA)
    k_gemm_xz<<<dim3(NXZ / 128, MQ / 128), 256, 0, stream>>>(ws_xm, ws_win, ws_xz);

    // depthwise conv + silu
    k_conv<<<dim3(DIQ / 256, MQ), 256, 0, stream>>>(ws_xz, conv_w, conv_b, ws_uf, ws_ub);

    // x_proj GEMM
    k_gemm_dbl<<<MQ / 128, 256, 0, stream>>>(ws_ub, ws_wxp, ws_dbl);

    // dt_proj + softplus
    k_delta<<<dim3(DIQ / 256, MQ), 256, 0, stream>>>(ws_dbl, dt_proj_w, dt_proj_b, ws_delta);

    // sequential selective scan + gating
    k_scan<<<BQ * (DIQ / 256), 256, 0, stream>>>(ws_delta, ws_uf, ws_dbl, A_log,
                                                 D_param, ws_xz, ws_yz);

    // out_proj GEMM + residual + gate (TDM + WMMA)
    k_gemm_out<<<dim3(DQ / 128, MQ / 128), 256, 0, stream>>>(ws_yz, ws_wout, x, ws_ada, out);

    (void)in_sizes; (void)n_in; (void)out_size; (void)ws_size;
}
